// CNNMNIST_49280454754490
// MI455X (gfx1250) — 8-bit matrix operand order repaired and checked by reading, compiled, not run on hardware
//
#include <hip/hip_runtime.h>
#include <hip/hip_bf16.h>


typedef int v8i_t __attribute__((ext_vector_type(8)));

#define EPSQ 1e-5f

__global__ void conv_wprep(const float* __restrict__ w1,
                           const float* __restrict__ w1b,
                           const float* __restrict__ w2,
                           float* __restrict__ q1,
                           float* __restrict__ q1b,
                           float* __restrict__ q2) {
  const float* src; float* dst; int n;
  if (blockIdx.x == 0)      { src = w1;  dst = q1;  n = 16 * 9;   }
  else if (blockIdx.x == 1) { src = w1b; dst = q1b; n = 16 * 9;   }
  else                      { src = w2;  dst = q2;  n = 96 * 144; }
  __shared__ float red[256];
  float a = 0.f;
  for (int i = threadIdx.x; i < n; i += 256) a += fabsf(src[i]);
  red[threadIdx.x] = a;
  __syncthreads();
  for (int s = 128; s > 0; s >>= 1) {
    if (threadIdx.x < s) red[threadIdx.x] += red[threadIdx.x + s];
    __syncthreads();
  }
  const float mag   = fmaxf(red[0] / (float)n, EPSQ);
  const float scale = 0.25f * 8.f / mag;
  for (int i = threadIdx.x; i < n; i += 256) {
    float q = fminf(fmaxf(rintf(src[i] * scale - 0.5f), -8.f), 7.f);
    dst[i] = (q + 0.5f) / scale;
  }
}

__global__ void fc_reduce(const float* __restrict__ w0, const float* __restrict__ w1,
                          const float* __restrict__ w2, const float* __restrict__ w3,
                          float* __restrict__ out  ) {
  const float* w; int n;
  switch (blockIdx.x) {
    case 0:  w = w0; n = 512 * 96;  break;
    case 1:  w = w1; n = 512 * 512; break;
    case 2:  w = w2; n = 512 * 512; break;
    default: w = w3; n = 10 * 512;  break;
  }
  __shared__ float rs[256], ra[256];
  float s = 0.f, a = 0.f;
  for (int i = threadIdx.x; i < n; i += 256) { float v = w[i]; s += v; a += fabsf(v); }
  rs[threadIdx.x] = s; ra[threadIdx.x] = a;
  __syncthreads();
  for (int st = 128; st > 0; st >>= 1) {
    if (threadIdx.x < st) {
      rs[threadIdx.x] += rs[threadIdx.x + st];
      ra[threadIdx.x] += ra[threadIdx.x + st];
    }
    __syncthreads();
  }
  if (threadIdx.x == 0) {
    out[blockIdx.x * 2 + 0] = rs[0] / (float)n;
    out[blockIdx.x * 2 + 1] = ra[0] / (float)n;
  }
}

__global__ void pack_binary(const float* __restrict__ w, const float* __restrict__ meanp,
                            signed char* __restrict__ dst, int Na, int Ka, int KT, int NT) {
  const int g = blockIdx.x * blockDim.x + threadIdx.x;
  const int total = NT * KT * 1024;
  if (g >= total) return;
  const float mean = meanp[0];
  const int tile = g >> 10;
  const int l = (g >> 5) & 31;
  const int d = g & 31;
  const int v = d >> 2, j = d & 3;
  const int nl = l & 15;
  const int kl = ((l >> 4) << 3) + ((v & 1) << 2) + j + ((v >> 1) << 4);
  const int ntile = tile / KT, ktile = tile - ntile * KT;
  const int n = ntile * 16 + nl, k = ktile * 64 + kl;
  signed char s = 0;
  if (n < Na && k < Ka) {
    float dd = w[(size_t)n * Ka + k] - mean;
    s = (dd > 0.f) ? (signed char)1 : ((dd < 0.f) ? (signed char)-1 : (signed char)0);
  }
  dst[g] = s;
}

__device__ inline void quant_row(float* p, int w) {
  float ma = 0.f;
  for (int i = 0; i < w; ++i) ma = fmaxf(ma, fabsf(p[i]));
  const float s   = 127.f / fmaxf(ma, EPSQ);
  const float inv = 1.f / s;
  for (int i = 0; i < w; ++i)
    p[i] = fminf(fmaxf(rintf(p[i] * s), -128.f), 127.f) * inv;
}

__global__ void __launch_bounds__(128) conv_fused(const float* __restrict__ x,
                                                  const float* __restrict__ wq1,
                                                  const float* __restrict__ wq1b,
                                                  const float* __restrict__ wq2,
                                                  float* __restrict__ h0) {
  __shared__ float sx[16 * 16];
  __shared__ float sh1[16 * 14 * 14];
  __shared__ float sh2[16 * 12 * 12];
  const int tid = threadIdx.x;
  const float* xb = x + (size_t)blockIdx.x * 256;
  sx[tid] = xb[tid];
  sx[tid + 128] = xb[tid + 128];
  __syncthreads();
  if (tid < 16) quant_row(sx + tid * 16, 16);
  __syncthreads();
  for (int o = tid; o < 16 * 14 * 14; o += 128) {
    const int oc = o / 196, rem = o % 196, oy = rem / 14, ox = rem % 14;
    float acc = 0.f;
    for (int ky = 0; ky < 3; ++ky)
      for (int kx = 0; kx < 3; ++kx)
        acc += sx[(oy + ky) * 16 + ox + kx] * wq1[oc * 9 + ky * 3 + kx];
    sh1[o] = fmaxf(acc, 0.f);
  }
  __syncthreads();
  for (int r = tid; r < 16 * 14; r += 128) quant_row(sh1 + r * 14, 14);
  __syncthreads();
  for (int o = tid; o < 16 * 12 * 12; o += 128) {
    const int c = o / 144, rem = o % 144, oy = rem / 12, ox = rem % 12;
    float acc = 0.f;
    for (int ky = 0; ky < 3; ++ky)
      for (int kx = 0; kx < 3; ++kx)
        acc += sh1[c * 196 + (oy + ky) * 14 + ox + kx] * wq1b[c * 9 + ky * 3 + kx];
    sh2[o] = fmaxf(acc, 0.f);
  }
  __syncthreads();
  for (int r = tid; r < 16 * 12; r += 128) quant_row(sh2 + r * 12, 12);
  __syncthreads();
  if (tid < 96) {
    const int g = tid / 6;
    const float* wp = wq2 + tid * 144;
    const float* ip = sh2 + g * 144;
    float acc = 0.f;
    for (int i = 0; i < 144; ++i) acc += ip[i] * wp[i];
    h0[(size_t)blockIdx.x * 96 + tid] = fmaxf(acc, 0.f);
  }
}

template <int KP, bool RELU, bool FULLN>
__global__ void __launch_bounds__(128) bitlinear_gemm(
    const float* __restrict__ X, const signed char* __restrict__ Wpk,
    const float* __restrict__ wscp, float* __restrict__ Y,
    int Ka, int Na, int Np) {
  constexpr int KT = KP / 64;
  constexpr int CH = KP / 32;
  __shared__ signed char sA[64 * KP];
  __shared__ float sRowf[64];
  const int tid  = threadIdx.x;
  const int wave = tid >> 5;
  const int lane = tid & 31;
  const int m0   = blockIdx.x * 64;
  const float wsc = wscp[0];

  for (int rr = 0; rr < 16; ++rr) {
    const int rloc = wave * 16 + rr;
    const float* xr = X + (size_t)(m0 + rloc) * Ka;
    const int k0 = lane * CH;
    float vals[CH];
    float ss = 0.f, ma = 0.f;
    for (int i = 0; i < CH; ++i) {
      const int k = k0 + i;
      const float v = (k < Ka) ? xr[k] : 0.f;
      vals[i] = v;
      ss += v * v;
      ma = fmaxf(ma, fabsf(v));
    }
    for (int off = 16; off >= 1; off >>= 1) {
      ss += __shfl_xor(ss, off, 32);
      ma = fmaxf(ma, __shfl_xor(ma, off, 32));
    }
    const float rs     = rsqrtf(ss / (float)Ka + EPSQ);
    const float ascale = 127.f / fmaxf(ma * rs, EPSQ);
    if (lane == 0) sRowf[rloc] = 1.f / ascale;
    unsigned int* dst = (unsigned int*)(sA + rloc * KP + k0);
    for (int i = 0; i < CH; i += 4) {
      unsigned int pk = 0;
      for (int j = 0; j < 4; ++j) {
        const float q = fminf(fmaxf(rintf(vals[i + j] * rs * ascale), -128.f), 127.f);
        pk |= ((unsigned int)((int)q & 0xFF)) << (8 * j);
      }
      dst[i >> 2] = pk;
    }
  }
  __syncthreads();

  const int half = lane >> 4;
  const int mr   = lane & 15;
  const int* sAi = (const int*)sA;
  v8i_t afr[KT];
  const int rowd = ((wave * 16 + mr) * KP) >> 2;
#pragma unroll
  for (int kt = 0; kt < KT; ++kt) {
    const int base = rowd + ((kt * 64 + half * 8) >> 2);
    v8i_t a;
    a[0] = sAi[base + 0];  a[1] = sAi[base + 1];
    a[2] = sAi[base + 4];  a[3] = sAi[base + 5];
    a[4] = sAi[base + 8];  a[5] = sAi[base + 9];
    a[6] = sAi[base + 12]; a[7] = sAi[base + 13];
    afr[kt] = a;
  }
  float rf[8];
#pragma unroll
  for (int i = 0; i < 8; ++i) rf[i] = sRowf[wave * 16 + half * 8 + i];

  const int NT = Np >> 4;
  for (int nt = 0; nt < NT; nt += 2) {
    if (nt + 2 < NT)
      __builtin_prefetch((const char*)Wpk + (size_t)(nt + 2) * KT * 1024 + lane * 64, 0, 1);
    const int4* bp0 = (const int4*)(Wpk + (size_t)nt * KT * 1024) + lane * 2;
    const int4* bp1 = bp0 + KT * 64;
    v8i_t acc0 = {0, 0, 0, 0, 0, 0, 0, 0};
    v8i_t acc1 = {0, 0, 0, 0, 0, 0, 0, 0};
    int4 p0a[2], p0b[2], p1a[2], p1b[2];
    p0a[0] = bp0[0]; p0b[0] = bp0[1];
    p1a[0] = bp1[0]; p1b[0] = bp1[1];
    if (KT > 1) {
      p0a[1] = bp0[64]; p0b[1] = bp0[65];
      p1a[1] = bp1[64]; p1b[1] = bp1[65];
    }
#pragma unroll
    for (int kt = 0; kt < KT; ++kt) {
      const int cur = kt & 1;
      v8i_t b0, b1;
      b0[0] = p0a[cur].x; b0[1] = p0a[cur].y; b0[2] = p0a[cur].z; b0[3] = p0a[cur].w;
      b0[4] = p0b[cur].x; b0[5] = p0b[cur].y; b0[6] = p0b[cur].z; b0[7] = p0b[cur].w;
      b1[0] = p1a[cur].x; b1[1] = p1a[cur].y; b1[2] = p1a[cur].z; b1[3] = p1a[cur].w;
      b1[4] = p1b[cur].x; b1[5] = p1b[cur].y; b1[6] = p1b[cur].z; b1[7] = p1b[cur].w;
      if (kt + 2 < KT) {
        const int o = (kt + 2) * 64;
        p0a[cur] = bp0[o]; p0b[cur] = bp0[o + 1];
        p1a[cur] = bp1[o]; p1b[cur] = bp1[o + 1];
      }
      acc0 = __builtin_amdgcn_wmma_i32_16x16x64_iu8(true, afr[kt], true, b0, acc0,
                                                    false, false);
      acc1 = __builtin_amdgcn_wmma_i32_16x16x64_iu8(true, afr[kt], true, b1, acc1,
                                                    false, false);
    }
    const int n0 = nt * 16 + mr;
    const int n1 = n0 + 16;
    float* yrow = Y + (size_t)(m0 + wave * 16 + half * 8) * Na;
    if (FULLN || n0 < Na) {
      float* y0 = yrow + n0;
#pragma unroll
      for (int i = 0; i < 8; ++i) {
        float v = (float)acc0[i] * rf[i] * wsc;
        if (RELU) v = fmaxf(v, 0.f);
        y0[(size_t)i * Na] = v;
      }
    }
    if (FULLN || n1 < Na) {
      float* y1 = yrow + n1;
#pragma unroll
      for (int i = 0; i < 8; ++i) {
        float v = (float)acc1[i] * rf[i] * wsc;
        if (RELU) v = fmaxf(v, 0.f);
        y1[(size_t)i * Na] = v;
      }
    }
  }
}

extern "C" void kernel_launch(void* const* d_in, const int* in_sizes, int n_in,
                              void* d_out, int out_size, void* d_ws, size_t ws_size,
                              hipStream_t stream) {
  (void)in_sizes; (void)n_in; (void)out_size; (void)ws_size;
  const float* x    = (const float*)d_in[0];
  const float* c1w  = (const float*)d_in[1];
  const float* c1bw = (const float*)d_in[2];
  const float* c2w  = (const float*)d_in[3];
  const float* fc1w = (const float*)d_in[4];
  const float* fc2w = (const float*)d_in[5];
  const float* fc3w = (const float*)d_in[6];
  const float* fclw = (const float*)d_in[7];
  float* out = (float*)d_out;
  char* ws   = (char*)d_ws;

  const int B = 16384;
  size_t off = 0;
  auto take = [&](size_t bytes) {
    size_t o = off;
    off += (bytes + 255) & ~(size_t)255;
    return o;
  };
  float* h0   = (float*)(ws + take((size_t)B * 96 * 4));
  float* bufA = (float*)(ws + take((size_t)B * 512 * 4));
  float* bufB = (float*)(ws + take((size_t)B * 512 * 4));
  float* wq1  = (float*)(ws + take(144 * 4));
  float* wq1b = (float*)(ws + take(144 * 4));
  float* wq2  = (float*)(ws + take(13824 * 4));
  signed char* pk1 = (signed char*)(ws + take(512 * 128));
  signed char* pk2 = (signed char*)(ws + take(512 * 512));
  signed char* pk3 = (signed char*)(ws + take(512 * 512));
  signed char* pkl = (signed char*)(ws + take(32 * 512));
  float* scales = (float*)(ws + take(8 * 4));

  conv_wprep<<<3, 256, 0, stream>>>(c1w, c1bw, c2w, wq1, wq1b, wq2);
  fc_reduce<<<4, 256, 0, stream>>>(fc1w, fc2w, fc3w, fclw, scales);
  pack_binary<<<(512 * 128) / 256, 256, 0, stream>>>(fc1w, scales + 0, pk1, 512, 96, 2, 32);
  pack_binary<<<(512 * 512) / 256, 256, 0, stream>>>(fc2w, scales + 2, pk2, 512, 512, 8, 32);
  pack_binary<<<(512 * 512) / 256, 256, 0, stream>>>(fc3w, scales + 4, pk3, 512, 512, 8, 32);
  pack_binary<<<(32 * 512) / 256, 256, 0, stream>>>(fclw, scales + 6, pkl, 10, 512, 8, 2);

  conv_fused<<<B, 128, 0, stream>>>(x, wq1, wq1b, wq2, h0);

  bitlinear_gemm<128, true,  true ><<<B / 64, 128, 0, stream>>>(h0,   pk1, scales + 1, bufA, 96,  512, 512);
  bitlinear_gemm<512, true,  true ><<<B / 64, 128, 0, stream>>>(bufA, pk2, scales + 3, bufB, 512, 512, 512);
  bitlinear_gemm<512, true,  true ><<<B / 64, 128, 0, stream>>>(bufB, pk3, scales + 5, bufA, 512, 512, 512);
  bitlinear_gemm<512, false, false><<<B / 64, 128, 0, stream>>>(bufA, pkl, scales + 7, out,  512, 10,  32);
}
